// ISAB_70970039599920
// MI455X (gfx1250) — matrix operand order repaired and checked by reading, not compiled, not run on hardware
//
#include <hip/hip_runtime.h>
#include <stdint.h>

// ISAB (Set Transformer) forward for MI455X / gfx1250.
// Big 512x512 projections: v_wmma_f32_16x16x32_f16 with double-buffered LDS
// staged by the CDNA5 async engine (global_load_async_to_lds_b128 + ASYNCcnt).
// Attention / softmax / LayerNorm epilogues in f32 VALU (Nq=9 or Nk=9).

typedef __attribute__((ext_vector_type(16))) _Float16 v16h;
typedef __attribute__((ext_vector_type(8)))  _Float16 v8h;
typedef __attribute__((ext_vector_type(8)))  float    v8f;

#define DMODEL 512
#define NHEAD  8
#define DHEAD  64
#define NBATCH 32
#define NSEQ   2048
#define NIND   9

// ---------------------------------------------------------------- async copy
// CDNA5 async global->LDS (no VGPR round-trip, tracked with ASYNCcnt).
__device__ __forceinline__ void async_copy_b128(unsigned lds_off,
                                                const _Float16* gsrc) {
  asm volatile("global_load_async_to_lds_b128 %0, %1, off"
               : : "v"(lds_off), "v"(gsrc) : "memory");
}
#define WAIT_ASYNCCNT(imm) asm volatile("s_wait_asynccnt " imm ::: "memory")

// ---------------------------------------------------------------- conversions
__global__ __launch_bounds__(256)
void k_cvt_f16(const float* __restrict__ in, _Float16* __restrict__ out, int n) {
  int i = blockIdx.x * blockDim.x + threadIdx.x;
  if (i < n) out[i] = (_Float16)in[i];
}

// W (k,n) f32 -> Wt (n,k) f16, 16x16 LDS tile transpose for coalescing
__global__ __launch_bounds__(256)
void k_wcvt_t(const float* __restrict__ W, _Float16* __restrict__ Wt) {
  __shared__ float t[16][17];
  int nb = blockIdx.x * 16, kb = blockIdx.y * 16;
  t[threadIdx.y][threadIdx.x] = W[(kb + threadIdx.y) * DMODEL + nb + threadIdx.x];
  __syncthreads();
  Wt[(size_t)(nb + threadIdx.y) * DMODEL + kb + threadIdx.x] =
      (_Float16)t[threadIdx.x][threadIdx.y];
}

// ---------------------------------------------------------------- WMMA GEMM
// C[M x 512] = A[M x 512] @ W + bias  with Wt = W^T (n,k) in f16.
// Block tile 128(M) x 128(N); 8 waves = 4 along M x 2 along N; each wave owns
// a 32x64 strip = 2 A-frags x 4 B-frags = 8 WMMAs per K-step (K stepped by 32).
// LDS double buffered; staged via async-to-LDS; s_wait_asynccnt 4 retires the
// current tile while next tile's 4 per-thread loads remain in flight.
#define TM 128
#define TN 128
#define TK 32

__global__ __launch_bounds__(256)
void k_gemm_f16(const _Float16* __restrict__ A,   // M x 512 row-major
                const _Float16* __restrict__ Bt,  // 512(n) x 512(k)
                const float*    __restrict__ bias,
                float*          __restrict__ Cf,  // M x 512 (nullable)
                _Float16*       __restrict__ Ch,  // M x 512 (nullable)
                int M)
{
  __shared__ _Float16 lA[2][TM][TK];   // 2 x 8 KB
  __shared__ _Float16 lB[2][TN][TK];   // 2 x 8 KB
  const int K = DMODEL, N = DMODEL;

  int tid  = threadIdx.x;
  int lane = tid & 31;
  int wave = tid >> 5;
  int mw = wave & 3;          // wave's M quarter (32 rows)
  int nw = wave >> 2;         // wave's N half   (64 cols)
  int m0 = blockIdx.x * TM;
  int n0 = blockIdx.y * TN;

  int row = lane & 15;        // row/col within a 16-wide fragment
  int grp = lane >> 4;        // lane half-group

  v8f acc[2][4] = {};

  // stage one K-tile (4 async b128 per thread: 2 for A strip, 2 for B strip)
  auto stage = [&](int sb, int kk) {
    int r = tid >> 1;               // 0..127
    int c = (tid & 1) * 16;         // 0 or 16 halves
    {
      int gr = m0 + r; if (gr >= M) gr = M - 1;   // clamp: OOB rows never stored
      const _Float16* src = A + (size_t)gr * K + kk + c;
      unsigned dst = (unsigned)(uintptr_t)&lA[sb][r][c];
      async_copy_b128(dst,      src);
      async_copy_b128(dst + 16, src + 8);
    }
    {
      const _Float16* src = Bt + (size_t)(n0 + r) * K + kk + c;
      unsigned dst = (unsigned)(uintptr_t)&lB[sb][r][c];
      async_copy_b128(dst,      src);
      async_copy_b128(dst + 16, src + 8);
    }
  };

  stage(0, 0);
  int buf = 0;
  for (int kb = 0; kb < K; kb += TK) {
    if (kb + TK < K) {
      stage(buf ^ 1, kb + TK);          // prefetch next tile into other buffer
      WAIT_ASYNCCNT("4");               // retire current tile's 4 loads
    } else {
      WAIT_ASYNCCNT("0");
    }
    __syncthreads();

    // load ALL fragments first, then back-to-back WMMAs
    v16h af[2], bf[4];
#pragma unroll
    for (int i = 0; i < 2; ++i) {
      const _Float16* ap = &lA[buf][mw * 32 + i * 16 + row][0];
      v8h lo = *(const v8h*)(ap + grp * 8);        // K = grp*8 .. +7
      v8h hi = *(const v8h*)(ap + 16 + grp * 8);   // K = 16+grp*8 .. +7
#pragma unroll
      for (int e = 0; e < 8; ++e) { af[i][e] = lo[e]; af[i][8 + e] = hi[e]; }
    }
#pragma unroll
    for (int j = 0; j < 4; ++j) {
      const _Float16* bp = &lB[buf][nw * 64 + j * 16 + row][0];
      v8h b0 = *(const v8h*)(bp + grp * 8);
      v8h b1 = *(const v8h*)(bp + 16 + grp * 8);
#pragma unroll
      for (int e = 0; e < 8; ++e) { bf[j][e] = b0[e]; bf[j][8 + e] = b1[e]; }
    }
#pragma unroll
    for (int i = 0; i < 2; ++i)
#pragma unroll
      for (int j = 0; j < 4; ++j)
        acc[i][j] = __builtin_amdgcn_wmma_f32_16x16x32_f16(
            false, af[i], false, bf[j], (short)0, acc[i][j], false, false);

    __syncthreads();                    // protect buffer before re-staging
    buf ^= 1;
  }

  // epilogue: C 16x16 layout -> lane col = row, VGPR r -> row grp*8+r
#pragma unroll
  for (int i = 0; i < 2; ++i) {
#pragma unroll
    for (int j = 0; j < 4; ++j) {
      int gcol = n0 + nw * 64 + j * 16 + row;
      float bv = bias ? bias[gcol] : 0.f;
#pragma unroll
      for (int r = 0; r < 8; ++r) {
        int grow = m0 + mw * 32 + i * 16 + grp * 8 + r;
        if (grow < M) {
          float v = acc[i][j][r] + bv;
          if (Cf) Cf[(size_t)grow * N + gcol] = v;
          if (Ch) Ch[(size_t)grow * N + gcol] = (_Float16)v;
        }
      }
    }
  }
}

// ---------------------------------------------------------------- MAB0 attn
// One block per (q, h, b): softmax over Nk=2048 keys, scores live in LDS.
__global__ __launch_bounds__(256)
void k_attn0(const float* __restrict__ Qp,   // 9 x 512 (shared by all batches)
             const float* __restrict__ Kp,   // B*2048 x 512
             const float* __restrict__ Vp,   // B*2048 x 512
             float* __restrict__ Oacc)       // B*9 x 512
{
  int q = blockIdx.x, h = blockIdx.y, b = blockIdx.z;
  __shared__ float sS[NSEQ];
  __shared__ float sQ[DHEAD];
  __shared__ float red[256];
  int tid = threadIdx.x;
  const float scale = 0.044194173824159216f;  // 1/sqrt(512)

  if (tid < DHEAD) sQ[tid] = Qp[q * DMODEL + h * DHEAD + tid];
  __syncthreads();

  const float* Kb = Kp + (size_t)b * NSEQ * DMODEL + h * DHEAD;
  float lmax = -1e30f;
  for (int k = tid; k < NSEQ; k += 256) {
    const float* kr = Kb + (size_t)k * DMODEL;
    float s = 0.f;
#pragma unroll
    for (int d = 0; d < DHEAD; ++d) s += sQ[d] * kr[d];
    s *= scale;
    sS[k] = s;
    lmax = fmaxf(lmax, s);
  }
  red[tid] = lmax; __syncthreads();
  for (int st = 128; st > 0; st >>= 1) {
    if (tid < st) red[tid] = fmaxf(red[tid], red[tid + st]);
    __syncthreads();
  }
  float mx = red[0]; __syncthreads();

  float lsum = 0.f;
  for (int k = tid; k < NSEQ; k += 256) {
    float e = __expf(sS[k] - mx); sS[k] = e; lsum += e;
  }
  red[tid] = lsum; __syncthreads();
  for (int st = 128; st > 0; st >>= 1) {
    if (tid < st) red[tid] += red[tid + st];
    __syncthreads();
  }
  float inv = 1.f / red[0]; __syncthreads();

  int d = tid & 63, stripe = tid >> 6;          // 4 k-stripes x 64 dims
  const float* Vb = Vp + (size_t)b * NSEQ * DMODEL + h * DHEAD + d;
  float part = 0.f;
  int k0 = stripe * (NSEQ / 4), k1 = k0 + NSEQ / 4;
  for (int k = k0; k < k1; ++k) part += sS[k] * Vb[(size_t)k * DMODEL];
  red[tid] = part; __syncthreads();
  if (tid < 64) {
    float o = (red[tid] + red[tid + 64] + red[tid + 128] + red[tid + 192]) * inv;
    Oacc[((size_t)b * NIND + q) * DMODEL + h * DHEAD + d] = o;
  }
}

// ---------------------------------------------------------------- MAB1 attn
// One thread per (b,row,head): Nk = 9 keys; K1/V1 of batch cached in LDS.
__global__ __launch_bounds__(256)
void k_attn1(const float* __restrict__ Qp,  // B*2048 x 512
             const float* __restrict__ K1,  // B*9 x 512
             const float* __restrict__ V1,  // B*9 x 512
             float* __restrict__ O)         // B*2048 x 512 (attn part only)
{
  int gid = blockIdx.x * 256 + threadIdx.x;
  int h  = gid & 7;
  int br = gid >> 3;               // b*2048 + row
  int b  = br / NSEQ;
  __shared__ float sK[NIND * DMODEL];
  __shared__ float sV[NIND * DMODEL];
  for (int i = threadIdx.x; i < NIND * DMODEL; i += 256) {
    sK[i] = K1[(size_t)b * NIND * DMODEL + i];
    sV[i] = V1[(size_t)b * NIND * DMODEL + i];
  }
  __syncthreads();

  const float scale = 0.044194173824159216f;
  const float* qp = Qp + (size_t)br * DMODEL + h * DHEAD;
  float qv[DHEAD];
#pragma unroll
  for (int d = 0; d < DHEAD; ++d) qv[d] = qp[d];

  float s[NIND]; float mx = -1e30f;
#pragma unroll
  for (int j = 0; j < NIND; ++j) {
    const float* kr = &sK[j * DMODEL + h * DHEAD];
    float a = 0.f;
#pragma unroll
    for (int d = 0; d < DHEAD; ++d) a += qv[d] * kr[d];
    s[j] = a * scale; mx = fmaxf(mx, s[j]);
  }
  float sum = 0.f;
#pragma unroll
  for (int j = 0; j < NIND; ++j) { s[j] = __expf(s[j] - mx); sum += s[j]; }
  float inv = 1.f / sum;

  float* op = O + (size_t)br * DMODEL + h * DHEAD;
  for (int d = 0; d < DHEAD; ++d) {
    float a = 0.f;
#pragma unroll
    for (int j = 0; j < NIND; ++j) a += s[j] * sV[j * DMODEL + h * DHEAD + d];
    op[d] = a * inv;
  }
}

// ---------------------------------------------------------------- add + LN
// out = LayerNorm(Xres[row % res_mod] + maybe_relu(Y[row])) * g + b
__global__ __launch_bounds__(256)
void k_addln(const float* __restrict__ Xres,
             const float* __restrict__ Y,
             const float* __restrict__ gamma,
             const float* __restrict__ beta,
             float*    __restrict__ outF,   // nullable
             _Float16* __restrict__ outH,   // nullable
             int res_mod, int do_relu)
{
  int row  = blockIdx.x;
  int rrow = res_mod ? (row % res_mod) : row;
  int tid  = threadIdx.x;
  __shared__ float red[256];

  float x0 = Xres[(size_t)rrow * DMODEL + tid];
  float x1 = Xres[(size_t)rrow * DMODEL + tid + 256];
  float y0 = Y[(size_t)row * DMODEL + tid];
  float y1 = Y[(size_t)row * DMODEL + tid + 256];
  if (do_relu) { y0 = fmaxf(y0, 0.f); y1 = fmaxf(y1, 0.f); }
  float v0 = x0 + y0, v1 = x1 + y1;

  red[tid] = v0 + v1; __syncthreads();
  for (int st = 128; st > 0; st >>= 1) {
    if (tid < st) red[tid] += red[tid + st];
    __syncthreads();
  }
  float mean = red[0] * (1.f / DMODEL); __syncthreads();
  float d0 = v0 - mean, d1 = v1 - mean;
  red[tid] = d0 * d0 + d1 * d1; __syncthreads();
  for (int st = 128; st > 0; st >>= 1) {
    if (tid < st) red[tid] += red[tid + st];
    __syncthreads();
  }
  float rstd = rsqrtf(red[0] * (1.f / DMODEL) + 1e-5f);
  float o0 = d0 * rstd * gamma[tid]       + beta[tid];
  float o1 = d1 * rstd * gamma[tid + 256] + beta[tid + 256];
  if (outF) { outF[(size_t)row * DMODEL + tid] = o0;
              outF[(size_t)row * DMODEL + tid + 256] = o1; }
  if (outH) { outH[(size_t)row * DMODEL + tid] = (_Float16)o0;
              outH[(size_t)row * DMODEL + tid + 256] = (_Float16)o1; }
}

// ---------------------------------------------------------------- launch
extern "C" void kernel_launch(void* const* d_in, const int* in_sizes, int n_in,
                              void* d_out, int out_size, void* d_ws, size_t ws_size,
                              hipStream_t stream) {
  const float* X   = (const float*)d_in[0];
  const float* I   = (const float*)d_in[1];
  // m0: Wq2 bq3 Wk4 bk5 Wv6 bv7 Wo8 bo9 g0=10 b0=11 g1=12 b1=13
  // m1: Wq14 bq15 Wk16 bk17 Wv18 bv19 Wo20 bo21 g0=22 b0=23 g1=24 b1=25
  const float* Wsrc[8] = { (const float*)d_in[2],  (const float*)d_in[4],
                           (const float*)d_in[6],  (const float*)d_in[8],
                           (const float*)d_in[14], (const float*)d_in[16],
                           (const float*)d_in[18], (const float*)d_in[20] };
  const float* b_q0 = (const float*)d_in[3],  *b_k0 = (const float*)d_in[5];
  const float* b_v0 = (const float*)d_in[7],  *b_o0 = (const float*)d_in[9];
  const float* g0_0 = (const float*)d_in[10], *bb0_0 = (const float*)d_in[11];
  const float* g1_0 = (const float*)d_in[12], *bb1_0 = (const float*)d_in[13];
  const float* b_q1 = (const float*)d_in[15], *b_k1 = (const float*)d_in[17];
  const float* b_v1 = (const float*)d_in[19], *b_o1 = (const float*)d_in[21];
  const float* g0_1 = (const float*)d_in[22], *bb0_1 = (const float*)d_in[23];
  const float* g1_1 = (const float*)d_in[24], *bb1_1 = (const float*)d_in[25];

  const size_t MBIG = (size_t)NBATCH * NSEQ;      // 65536 rows
  const size_t MSML = (size_t)NBATCH * NIND;      // 288 rows

  // bump allocator over d_ws with region reuse
  char* ws = (char*)d_ws;
  size_t off = 0;
  auto alloc = [&](size_t bytes) -> char* {
    char* p = ws + off;
    off += (bytes + 255) & ~(size_t)255;
    return p;
  };
  _Float16* Xh    = (_Float16*)alloc(MBIG * DMODEL * 2);   // X in f16; reused as Oh1
  _Float16* Ih    = (_Float16*)alloc((size_t)NIND * DMODEL * 2);
  _Float16* Wt    = (_Float16*)alloc((size_t)8 * DMODEL * DMODEL * 2);
  float*    Qp0f  = (float*)alloc((size_t)NIND * DMODEL * 4);
  float*    Oacc0 = (float*)alloc(MSML * DMODEL * 4);
  float*    Opre0 = (float*)alloc(MSML * DMODEL * 4);
  _Float16* Opre0h= (_Float16*)alloc(MSML * DMODEL * 2);
  float*    R0f   = (float*)alloc(MSML * DMODEL * 4);
  float*    Hf    = (float*)alloc(MSML * DMODEL * 4);
  _Float16* Hh    = (_Float16*)alloc(MSML * DMODEL * 2);
  float*    K1f   = (float*)alloc(MSML * DMODEL * 4);
  float*    V1f   = (float*)alloc(MSML * DMODEL * 4);
  float*    bigA  = (float*)alloc(MBIG * DMODEL * 4);      // Kp0 -> Qp1 -> R1
  float*    bigB  = (float*)alloc(MBIG * DMODEL * 4);      // Vp0 -> attn1 out
  float*    bigC  = (float*)alloc(MBIG * DMODEL * 4);      // post-attn1 LN (f32)
  _Float16* Oh1   = Xh;                                    // Xh dead after Qp1 gemm

  float* out = (float*)d_out;

  const int nX = (int)(MBIG * DMODEL);
  const int nI = NIND * DMODEL;

  // 1) stage activations/weights in f16 (weights transposed)
  k_cvt_f16<<<(nX + 255) / 256, 256, 0, stream>>>(X, Xh, nX);
  k_cvt_f16<<<(nI + 255) / 256, 256, 0, stream>>>(I, Ih, nI);
  for (int w = 0; w < 8; ++w)
    k_wcvt_t<<<dim3(DMODEL / 16, DMODEL / 16), dim3(16, 16), 0, stream>>>(
        Wsrc[w], Wt + (size_t)w * DMODEL * DMODEL);
  _Float16* Wq0t = Wt + 0 * (size_t)DMODEL * DMODEL;
  _Float16* Wk0t = Wt + 1 * (size_t)DMODEL * DMODEL;
  _Float16* Wv0t = Wt + 2 * (size_t)DMODEL * DMODEL;
  _Float16* Wo0t = Wt + 3 * (size_t)DMODEL * DMODEL;
  _Float16* Wq1t = Wt + 4 * (size_t)DMODEL * DMODEL;
  _Float16* Wk1t = Wt + 5 * (size_t)DMODEL * DMODEL;
  _Float16* Wv1t = Wt + 6 * (size_t)DMODEL * DMODEL;
  _Float16* Wo1t = Wt + 7 * (size_t)DMODEL * DMODEL;

  dim3 gBig((unsigned)((MBIG + TM - 1) / TM), DMODEL / TN);
  dim3 gSml((unsigned)((MSML + TM - 1) / TM), DMODEL / TN);
  dim3 gOne(1, DMODEL / TN);

  // 2) MAB0 projections
  k_gemm_f16<<<gOne, 256, 0, stream>>>(Ih, Wq0t, b_q0, Qp0f, (_Float16*)nullptr, NIND);
  k_gemm_f16<<<gBig, 256, 0, stream>>>(Xh, Wk0t, b_k0, bigA, (_Float16*)nullptr, (int)MBIG);
  k_gemm_f16<<<gBig, 256, 0, stream>>>(Xh, Wv0t, b_v0, bigB, (_Float16*)nullptr, (int)MBIG);

  // 3) MAB0 attention + residual LN + FFN + LN  ->  H (B,9,512)
  k_attn0<<<dim3(NIND, NHEAD, NBATCH), 256, 0, stream>>>(Qp0f, bigA, bigB, Oacc0);
  k_addln<<<(unsigned)MSML, 256, 0, stream>>>(Qp0f, Oacc0, g0_0, bb0_0,
                                              Opre0, Opre0h, NIND, 0);
  k_gemm_f16<<<gSml, 256, 0, stream>>>(Opre0h, Wo0t, b_o0, R0f, (_Float16*)nullptr, (int)MSML);
  k_addln<<<(unsigned)MSML, 256, 0, stream>>>(Opre0, R0f, g1_0, bb1_0,
                                              Hf, Hh, 0, 1);

  // 4) MAB1 projections
  k_gemm_f16<<<gSml, 256, 0, stream>>>(Hh, Wk1t, b_k1, K1f, (_Float16*)nullptr, (int)MSML);
  k_gemm_f16<<<gSml, 256, 0, stream>>>(Hh, Wv1t, b_v1, V1f, (_Float16*)nullptr, (int)MSML);
  k_gemm_f16<<<gBig, 256, 0, stream>>>(Xh, Wq1t, b_q1, bigA, (_Float16*)nullptr, (int)MBIG);

  // 5) MAB1 attention (Nk = 9) + residual LN -> bigC(f32) + Oh1(f16)
  k_attn1<<<(unsigned)(MBIG * NHEAD / 256), 256, 0, stream>>>(bigA, K1f, V1f, bigB);
  k_addln<<<(unsigned)MBIG, 256, 0, stream>>>(bigA, bigB, g0_1, bb0_1,
                                              bigC, Oh1, 0, 0);

  // 6) FFN + final LN -> d_out
  k_gemm_f16<<<gBig, 256, 0, stream>>>(Oh1, Wo1t, b_o1, bigA, (_Float16*)nullptr, (int)MBIG);
  k_addln<<<(unsigned)MBIG, 256, 0, stream>>>(bigC, bigA, g1_1, bb1_1,
                                              out, (_Float16*)nullptr, 0, 1);
}
